// MetaSANRec_68564857914056
// MI455X (gfx1250) — matrix operand order repaired and checked by reading, not compiled, not run on hardware
//
#include <hip/hip_runtime.h>
#include <hip/hip_bf16.h>

// ---- problem constants --------------------------------------------------
#define B_   256
#define L_   200
#define LP   208          // L padded to 13 * 16 WMMA M-tiles
#define D_   256
#define H_   8
#define DK   32
#define IN_  64
#define HID  128
#define LAM  0.01f
#define RSQRT_DK 0.17677669529663687f   // 1/sqrt(32)

typedef __attribute__((ext_vector_type(16))) __bf16 v16bf;
typedef __attribute__((ext_vector_type(8)))  float  v8f;
typedef __attribute__((address_space(3))) __bf16 lds_bf16_t;

#define WMMA_BF16(a, b, c) \
  __builtin_amdgcn_wmma_f32_16x16x32_bf16(false, (a), false, (b), (short)0, (c), false, false)

// ---- async global -> LDS 16-byte copy (CDNA5, ASYNCcnt-tracked) ---------
__device__ __forceinline__ void async_cp16(void* lds_dst, const void* gsrc) {
  unsigned loff = (unsigned)(size_t)(lds_bf16_t*)lds_dst;   // LDS byte offset
  asm volatile("global_load_async_to_lds_b128 %0, %1, off"
               :: "v"(loff), "v"(gsrc) : "memory");
}
__device__ __forceinline__ void async_wait0() {
  asm volatile("s_wait_asynccnt 0" ::: "memory");
}

// ---- fragment helpers (wave32; A/B in contiguous k order, not ISA table order) ----------------------
// A (16x32 bf16): lane holds row M = lane%16; K(e) = 16*(lane/16) + e
__device__ __forceinline__ int a_koff(int e, int lane) {
  return ((lane >> 4) << 4) + e;
}
// C/D (16x16 f32): element i -> row = i + 8*(lane/16), col = lane%16
__device__ __forceinline__ int c_row(int i, int lane) { return i + ((lane >> 4) << 3); }

__device__ __forceinline__ v16bf load_a_bf16(const __bf16* rowptr, int k0, int lane) {
  v16bf a;
#pragma unroll
  for (int e = 0; e < 16; ++e) a[e] = rowptr[k0 + a_koff(e, lane)];
  return a;
}
// B (32x16) from TRANSPOSED bf16 weights laid out [N][K]:
// lane holds col N = col; K = k0 + 16*(lane/16) + i -> 16 contiguous bf16 (32B)
__device__ __forceinline__ v16bf load_bT(const __bf16* baseT, int ldT, int k0, int col, int lane) {
  const __bf16* p = baseT + (size_t)col * ldT + k0 + ((lane >> 4) << 4);
  v16bf b;
#pragma unroll
  for (int i = 0; i < 16; ++i) b[i] = p[i];
  return b;
}

// ---- K1: hypernetwork h1 = sigmoid(ug @ fc1_w + fc1_b) ------------------
__global__ void hyper_kernel(const float* __restrict__ ug, const float* __restrict__ fc1_w,
                             const float* __restrict__ fc1_b, __bf16* __restrict__ h1bf) {
  int b = blockIdx.x, j = threadIdx.x;      // 256 blocks x 128 threads
  float acc = fc1_b[j];
#pragma unroll 8
  for (int i = 0; i < IN_; ++i) acc += ug[b * IN_ + i] * fc1_w[i * HID + j];
  float s = 1.0f / (1.0f + __expf(-acc));
  h1bf[b * HID + j] = (__bf16)s;
}

// ---- K2: x (fp32, L=200) -> bf16 padded to LP=208 -----------------------
__global__ void cvt_x_kernel(const float* __restrict__ x, __bf16* __restrict__ xbf) {
  int idx = blockIdx.x * 256 + threadIdx.x;           // B*LP*D elements
  int d = idx & (D_ - 1);
  int l = (idx >> 8) % LP;
  int b = idx / (LP * D_);
  float v = (l < L_) ? x[((size_t)b * L_ + l) * D_ + d] : 0.0f;
  xbf[idx] = (__bf16)v;
}

// ---- K3: transpose+convert fp32 [K][N] -> bf16 [N][K] -------------------
__global__ void tpose_kernel(const float* __restrict__ in, __bf16* __restrict__ outT,
                             int kshift, int N) {
  int idx = blockIdx.x * 256 + threadIdx.x;   // total N << kshift
  int K = 1 << kshift;
  int k = idx & (K - 1);
  int n = idx >> kshift;
  outT[idx] = (__bf16)in[(size_t)k * N + n];
}

// ---- K4: w1T[b][e][d] = LAM*(h1@fc2_w + fc2_b) + base_w  (bf16, transposed)
// GEMM M=256(batch) K=128 N=65536 (n = d*256+e). grid(256,16), 8 waves x 2 tiles.
__global__ void w1_kernel(const __bf16* __restrict__ h1bf, const __bf16* __restrict__ fc2T,
                          const float* __restrict__ fc2_b, const float* __restrict__ base_w,
                          __bf16* __restrict__ w1T) {
  int lane = threadIdx.x & 31, wv = threadIdx.x >> 5;
  int n0 = blockIdx.x * 256 + wv * 32;
  int m0 = blockIdx.y * 16;
  v8f c0 = {}, c1 = {};
  const __bf16* arow = h1bf + (size_t)(m0 + (lane & 15)) * HID;
#pragma unroll
  for (int kk = 0; kk < HID; kk += 32) {
    v16bf a  = load_a_bf16(arow, kk, lane);
    v16bf b0 = load_bT(fc2T, HID, kk, n0 + (lane & 15), lane);
    v16bf b1 = load_bT(fc2T, HID, kk, n0 + 16 + (lane & 15), lane);
    c0 = WMMA_BF16(a, b0, c0);
    c1 = WMMA_BF16(a, b1, c1);
  }
#pragma unroll
  for (int i = 0; i < 8; ++i) {
    int m  = m0 + c_row(i, lane);
    int na = n0 + (lane & 15), nb = na + 16;
    // n = d*256 + e ; transposed store at [b][e][d]
    w1T[(size_t)m * (D_ * D_) + (na & 255) * D_ + (na >> 8)] =
        (__bf16)(LAM * (c0[i] + fc2_b[na]) + base_w[na]);
    w1T[(size_t)m * (D_ * D_) + (nb & 255) * D_ + (nb >> 8)] =
        (__bf16)(LAM * (c1[i] + fc2_b[nb]) + base_w[nb]);
  }
}

// ---- A-tile staging: 16 rows x 256 cols of bf16 into padded LDS ---------
// Padded ld = D_+8 -> row stride 528B (16B aligned, conflict-free frag reads)
#define XSH_LD (D_ + 8)
__device__ __forceinline__ void stage_a_tile(__bf16 (*xsh)[XSH_LD], const __bf16* src0,
                                             int tid) {
  for (int t = tid; t < 16 * 32; t += 256) {        // 512 chunks of 8 bf16
    int r = t >> 5, c = (t & 31) * 8;
    const __bf16* s = src0 + (size_t)r * D_ + c;
#pragma unroll
    for (int u = 0; u < 8; ++u) xsh[r][c + u] = s[u];
  }
  __syncthreads();
}

// ---- K5/K6: k = x@Wk + bk (row-major out), v = x@Wv + bv (headwise-T out)
// grid(13, 256), 8 waves x 32 cols = 256 cols
__global__ void proj_sharedw_kernel(const __bf16* __restrict__ xin, const __bf16* __restrict__ WT,
                                    const float* __restrict__ bias,
                                    __bf16* __restrict__ out, __bf16* __restrict__ outT) {
  __shared__ __bf16 xsh[16][XSH_LD];
  int lane = threadIdx.x & 31, wv = threadIdx.x >> 5;
  int mt = blockIdx.x, b = blockIdx.y;
  int n0 = wv * 32;
  stage_a_tile(xsh, xin + ((size_t)b * LP + mt * 16) * D_, threadIdx.x);
  const __bf16* arow = &xsh[lane & 15][0];
  v8f c0 = {}, c1 = {};
#pragma unroll
  for (int kk = 0; kk < D_; kk += 32) {
    v16bf a  = load_a_bf16(arow, kk, lane);
    v16bf b0 = load_bT(WT, D_, kk, n0 + (lane & 15), lane);
    v16bf b1 = load_bT(WT, D_, kk, n0 + 16 + (lane & 15), lane);
    c0 = WMMA_BF16(a, b0, c0);
    c1 = WMMA_BF16(a, b1, c1);
  }
#pragma unroll
  for (int i = 0; i < 8; ++i) {
    int row = mt * 16 + c_row(i, lane);
    int na = n0 + (lane & 15), nb = na + 16;
    __bf16 va = (__bf16)(c0[i] + bias[na]);
    __bf16 vb = (__bf16)(c1[i] + bias[nb]);
    if (out) {
      size_t base = ((size_t)b * LP + row) * D_;
      out[base + na] = va;
      out[base + nb] = vb;
    }
    if (outT) {   // [b][h][d][l] for attention P@V B-fragments
      outT[(((size_t)b * H_ + (na >> 5)) * DK + (na & 31)) * LP + row] = va;
      outT[(((size_t)b * H_ + (nb >> 5)) * DK + (nb & 31)) * LP + row] = vb;
    }
  }
}

// ---- K7: q = x @ w1[b] + b_q  (per-sample transposed bf16 weights) ------
__global__ void qproj_kernel(const __bf16* __restrict__ xin, const __bf16* __restrict__ w1T,
                             const float* __restrict__ b_q, __bf16* __restrict__ qout) {
  __shared__ __bf16 xsh[16][XSH_LD];
  int lane = threadIdx.x & 31, wv = threadIdx.x >> 5;
  int mt = blockIdx.x, b = blockIdx.y;
  int n0 = wv * 32;
  stage_a_tile(xsh, xin + ((size_t)b * LP + mt * 16) * D_, threadIdx.x);
  const __bf16* arow  = &xsh[lane & 15][0];
  const __bf16* wbase = w1T + (size_t)b * D_ * D_;   // [e][d]
  v8f c0 = {}, c1 = {};
#pragma unroll
  for (int kk = 0; kk < D_; kk += 32) {
    __builtin_prefetch((const void*)(wbase + (size_t)(n0 + (lane & 15)) * D_ + kk + 32), 0, 0);
    v16bf a  = load_a_bf16(arow, kk, lane);
    v16bf b0 = load_bT(wbase, D_, kk, n0 + (lane & 15), lane);
    v16bf b1 = load_bT(wbase, D_, kk, n0 + 16 + (lane & 15), lane);
    c0 = WMMA_BF16(a, b0, c0);
    c1 = WMMA_BF16(a, b1, c1);
  }
#pragma unroll
  for (int i = 0; i < 8; ++i) {
    int row = mt * 16 + c_row(i, lane);
    int na = n0 + (lane & 15), nb = na + 16;
    size_t base = ((size_t)b * LP + row) * D_;
    qout[base + na] = (__bf16)(c0[i] + b_q[na]);
    qout[base + nb] = (__bf16)(c1[i] + b_q[nb]);
  }
}

// ---- K8: causal flash attention per (b,h), dk=32, paired key tiles ------
// grid(H, B), 256 thr = 8 waves. K/V head slices staged in LDS via async DMA.
#define KSH_LD (DK + 8)     // 80B rows: 16B aligned, conflict-free
#define VSH_LD (LP + 8)     // 432B rows: 16B aligned, conflict-free
#define PB_LD  40           // pbuf row pad
__global__ void attn_kernel(const __bf16* __restrict__ qbf, const __bf16* __restrict__ kbf,
                            const __bf16* __restrict__ vT, __bf16* __restrict__ obf) {
  __shared__ __bf16 ksh[LP][KSH_LD];        // [key][d]
  __shared__ __bf16 vsh[DK][VSH_LD];        // [d][key]
  __shared__ __bf16 pbuf[8][16][PB_LD];
  int lane = threadIdx.x & 31, wv = threadIdx.x >> 5;
  int h = blockIdx.x, b = blockIdx.y;
  size_t hb  = (size_t)b * LP * D_ + h * DK;           // q/k row-major base
  size_t vtb = ((size_t)b * H_ + h) * DK * LP;         // vT [d][l] base

  // stage K head [208][32] and V head [32][208] via async global->LDS
  for (int t = threadIdx.x; t < LP * 4; t += 256) {    // 4 x 16B per key row
    int l = t >> 2, c = (t & 3) * 8;
    async_cp16(&ksh[l][c], kbf + hb + (size_t)l * D_ + c);
  }
  for (int t = threadIdx.x; t < DK * 26; t += 256) {   // 26 x 16B per d row
    int d = t / 26, c = (t % 26) * 8;
    async_cp16(&vsh[d][c], vT + vtb + (size_t)d * LP + c);
  }
  async_wait0();
  __syncthreads();

  for (int mt = wv; mt < LP / 16; mt += 8) {
    // Q A-fragment: 16 rows x K=32 (== dk), straight from global
    v16bf qa;
    {
      size_t qrow = hb + (size_t)(mt * 16 + (lane & 15)) * D_;
#pragma unroll
      for (int e = 0; e < 16; ++e) qa[e] = qbf[qrow + a_koff(e, lane)];
    }
    v8f o0 = {}, o1 = {};
    float rmax[8], rsum[8];
#pragma unroll
    for (int i = 0; i < 8; ++i) { rmax[i] = -1e9f; rsum[i] = 0.0f; }

    for (int j2 = 0; j2 <= mt; j2 += 2) {
      const bool has2 = (j2 + 1 <= mt);
      const int kk0 = ((lane >> 4) << 4);              // 0 or 16
      // B fragments = K^T for tiles j2, j2+1 (from LDS)
      v16bf kb0, kb1;
#pragma unroll
      for (int i = 0; i < 16; ++i) kb0[i] = ksh[j2 * 16 + (lane & 15)][kk0 + i];
      if (has2) {
#pragma unroll
        for (int i = 0; i < 16; ++i) kb1[i] = ksh[(j2 + 1) * 16 + (lane & 15)][kk0 + i];
      } else {
#pragma unroll
        for (int i = 0; i < 16; ++i) kb1[i] = (__bf16)0.0f;
      }
      v8f s0 = {}, s1 = {};
      s0 = WMMA_BF16(qa, kb0, s0);
      s1 = WMMA_BF16(qa, kb1, s1);

      // online softmax over the 32-key strip
#pragma unroll
      for (int i = 0; i < 8; ++i) {
        int rowg = mt * 16 + c_row(i, lane);
        int col0 = j2 * 16 + (lane & 15);
        int col1 = col0 + 16;
        float sv0 = s0[i] * RSQRT_DK;
        float sv1 = s1[i] * RSQRT_DK;
        if (col0 > rowg || col0 >= L_) sv0 = -1e9f;
        if (!has2 || col1 > rowg || col1 >= L_) sv1 = -1e9f;
        float mj = fmaxf(sv0, sv1);
        mj = fmaxf(mj, __shfl_xor(mj, 1, 32));
        mj = fmaxf(mj, __shfl_xor(mj, 2, 32));
        mj = fmaxf(mj, __shfl_xor(mj, 4, 32));
        mj = fmaxf(mj, __shfl_xor(mj, 8, 32));
        float mnew  = fmaxf(rmax[i], mj);
        float scale = __expf(rmax[i] - mnew);
        float p0    = __expf(sv0 - mnew);
        float p1    = __expf(sv1 - mnew);
        float ps = p0 + p1;
        ps += __shfl_xor(ps, 1, 32);
        ps += __shfl_xor(ps, 2, 32);
        ps += __shfl_xor(ps, 4, 32);
        ps += __shfl_xor(ps, 8, 32);
        rsum[i] = rsum[i] * scale + ps;
        rmax[i] = mnew;
        o0[i] *= scale; o1[i] *= scale;
        pbuf[wv][c_row(i, lane)][lane & 15]        = (__bf16)p0;   // C-layout -> LDS
        pbuf[wv][c_row(i, lane)][16 + (lane & 15)] = (__bf16)p1;
      }
      asm volatile("s_wait_dscnt 0" ::: "memory");   // same-wave LDS RAW

      // P as A-fragment: full K=32 (two key tiles)
      v16bf pa;
      {
        int m = lane & 15;
#pragma unroll
        for (int e = 0; e < 16; ++e) pa[e] = pbuf[wv][m][a_koff(e, lane)];
      }
      // V B-fragments (from LDS, contiguous along key index)
      v16bf vb0, vb1;
      if ((kk0 == 0) || has2) {
#pragma unroll
        for (int i = 0; i < 16; ++i) {
          vb0[i] = vsh[lane & 15][j2 * 16 + kk0 + i];
          vb1[i] = vsh[16 + (lane & 15)][j2 * 16 + kk0 + i];
        }
      } else {
#pragma unroll
        for (int i = 0; i < 16; ++i) { vb0[i] = (__bf16)0.0f; vb1[i] = (__bf16)0.0f; }
      }
      o0 = WMMA_BF16(pa, vb0, o0);
      o1 = WMMA_BF16(pa, vb1, o1);
    }

#pragma unroll
    for (int i = 0; i < 8; ++i) {
      int rowg = mt * 16 + c_row(i, lane);
      float inv = 1.0f / (rsum[i] + 1e-20f);
      size_t base = hb + (size_t)rowg * D_;
      obf[base + (lane & 15)]      = (__bf16)(o0[i] * inv);
      obf[base + 16 + (lane & 15)] = (__bf16)(o1[i] * inv);
    }
  }
}

// ---- K9: out = o @ Wo + bo (fp32 store, rows < 200 only) ----------------
__global__ void out_kernel(const __bf16* __restrict__ obf, const __bf16* __restrict__ WoT,
                           const float* __restrict__ bo, float* __restrict__ out) {
  __shared__ __bf16 xsh[16][XSH_LD];
  int lane = threadIdx.x & 31, wv = threadIdx.x >> 5;
  int mt = blockIdx.x, b = blockIdx.y;
  int n0 = wv * 32;
  stage_a_tile(xsh, obf + ((size_t)b * LP + mt * 16) * D_, threadIdx.x);
  const __bf16* arow = &xsh[lane & 15][0];
  v8f c0 = {}, c1 = {};
#pragma unroll
  for (int kk = 0; kk < D_; kk += 32) {
    v16bf a  = load_a_bf16(arow, kk, lane);
    v16bf b0 = load_bT(WoT, D_, kk, n0 + (lane & 15), lane);
    v16bf b1 = load_bT(WoT, D_, kk, n0 + 16 + (lane & 15), lane);
    c0 = WMMA_BF16(a, b0, c0);
    c1 = WMMA_BF16(a, b1, c1);
  }
#pragma unroll
  for (int i = 0; i < 8; ++i) {
    int row = mt * 16 + c_row(i, lane);
    if (row < L_) {
      int na = n0 + (lane & 15), nb = na + 16;
      size_t base = ((size_t)b * L_ + row) * D_;
      out[base + na] = c0[i] + bo[na];
      out[base + nb] = c1[i] + bo[nb];
    }
  }
}

// ---- launch -------------------------------------------------------------
extern "C" void kernel_launch(void* const* d_in, const int* in_sizes, int n_in,
                              void* d_out, int out_size, void* d_ws, size_t ws_size,
                              hipStream_t stream) {
  (void)in_sizes; (void)n_in; (void)out_size; (void)ws_size;
  const float* x      = (const float*)d_in[0];
  const float* ug     = (const float*)d_in[1];
  const float* fc1_w  = (const float*)d_in[2];
  const float* fc1_b  = (const float*)d_in[3];
  const float* fc2_w  = (const float*)d_in[4];
  const float* fc2_b  = (const float*)d_in[5];
  const float* base_w = (const float*)d_in[6];
  const float* b_q    = (const float*)d_in[7];
  const float* Wk     = (const float*)d_in[8];
  const float* bk     = (const float*)d_in[9];
  const float* Wv     = (const float*)d_in[10];
  const float* bv     = (const float*)d_in[11];
  const float* Wo     = (const float*)d_in[12];
  const float* bo     = (const float*)d_in[13];
  float* out = (float*)d_out;

  char* ws = (char*)d_ws;
  size_t off = 0;
  const size_t sz_tok = (size_t)B_ * LP * D_ * sizeof(__bf16);      // 27.3 MB
  const size_t sz_w   = (size_t)D_ * D_ * sizeof(__bf16);           // 128 KB
  __bf16* h1bf = (__bf16*)(ws + off); off += (size_t)B_ * HID * sizeof(__bf16);
  off = (off + 255) & ~(size_t)255;
  __bf16* xbf  = (__bf16*)(ws + off); off += sz_tok;
  __bf16* fc2T = (__bf16*)(ws + off); off += (size_t)HID * D_ * D_ * sizeof(__bf16);
  __bf16* WkT  = (__bf16*)(ws + off); off += sz_w;
  __bf16* WvT  = (__bf16*)(ws + off); off += sz_w;
  __bf16* WoT  = (__bf16*)(ws + off); off += sz_w;
  __bf16* w1T  = (__bf16*)(ws + off); off += (size_t)B_ * D_ * D_ * sizeof(__bf16);
  __bf16* qbf  = (__bf16*)(ws + off); off += sz_tok;
  __bf16* kbf  = (__bf16*)(ws + off); off += sz_tok;
  __bf16* vT   = (__bf16*)(ws + off); off += sz_tok;                // [b][h][d][l]
  __bf16* obf  = (__bf16*)(ws + off); off += sz_tok;

  hyper_kernel<<<dim3(B_), dim3(HID), 0, stream>>>(ug, fc1_w, fc1_b, h1bf);
  cvt_x_kernel<<<dim3((B_ * LP * D_) / 256), dim3(256), 0, stream>>>(x, xbf);
  tpose_kernel<<<dim3((HID * D_ * D_) / 256), dim3(256), 0, stream>>>(fc2_w, fc2T, 7, D_ * D_);
  tpose_kernel<<<dim3((D_ * D_) / 256), dim3(256), 0, stream>>>(Wk, WkT, 8, D_);
  tpose_kernel<<<dim3((D_ * D_) / 256), dim3(256), 0, stream>>>(Wv, WvT, 8, D_);
  tpose_kernel<<<dim3((D_ * D_) / 256), dim3(256), 0, stream>>>(Wo, WoT, 8, D_);
  w1_kernel<<<dim3((D_ * D_) / 256, B_ / 16), dim3(256), 0, stream>>>(h1bf, fc2T, fc2_b, base_w, w1T);
  proj_sharedw_kernel<<<dim3(LP / 16, B_), dim3(256), 0, stream>>>(xbf, WkT, bk, kbf, nullptr);
  proj_sharedw_kernel<<<dim3(LP / 16, B_), dim3(256), 0, stream>>>(xbf, WvT, bv, nullptr, vT);
  qproj_kernel<<<dim3(LP / 16, B_), dim3(256), 0, stream>>>(xbf, w1T, b_q, qbf);
  attn_kernel<<<dim3(H_, B_), dim3(256), 0, stream>>>(qbf, kbf, vT, obf);
  out_kernel<<<dim3(LP / 16, B_), dim3(256), 0, stream>>>(obf, WoT, bo, out);
}
